// MultiQueryAttention_73813307949346
// MI455X (gfx1250) — matrix operand order repaired and checked by reading, not compiled, not run on hardware
//
#include <hip/hip_runtime.h>

typedef __attribute__((ext_vector_type(16))) __bf16 v16bf;
typedef __attribute__((ext_vector_type(8)))  float  v8f;

union FragAB { v16bf v; uint4 q[2]; unsigned short h[16]; };
union FragC  { v8f   v; float f[8]; };

__device__ inline unsigned short f2bf(float x) {
  union { float f; unsigned u; } c; c.f = x;
  unsigned r = c.u + 0x7FFFu + ((c.u >> 16) & 1u);
  return (unsigned short)(r >> 16);
}

__device__ inline uint4 pack8bf(const float* v) {
  uint4 r;
  r.x = (unsigned)f2bf(v[0]) | ((unsigned)f2bf(v[1]) << 16);
  r.y = (unsigned)f2bf(v[2]) | ((unsigned)f2bf(v[3]) << 16);
  r.z = (unsigned)f2bf(v[4]) | ((unsigned)f2bf(v[5]) << 16);
  r.w = (unsigned)f2bf(v[6]) | ((unsigned)f2bf(v[7]) << 16);
  return r;
}

// ---- CDNA5 async global->LDS staging (ASYNCcnt path), with portable fallback ----
#if __has_builtin(__builtin_amdgcn_global_load_async_to_lds_b128)
#define MQA_USE_ASYNC 1
#else
#define MQA_USE_ASYNC 0
#endif

typedef int v4i __attribute__((vector_size(16)));
typedef __attribute__((address_space(1))) v4i gv4i;  // global
typedef __attribute__((address_space(3))) v4i lv4i;  // LDS

__device__ inline void async_ld16(unsigned short* lds, const unsigned short* g) {
#if MQA_USE_ASYNC
  __builtin_amdgcn_global_load_async_to_lds_b128((gv4i*)g, (lv4i*)lds, 0, 0);
#else
  *(uint4*)lds = *(const uint4*)g;
#endif
}

__device__ inline void async_wait() {
#if MQA_USE_ASYNC
#if __has_builtin(__builtin_amdgcn_s_wait_asynccnt)
  __builtin_amdgcn_s_wait_asynccnt(0);
#else
  asm volatile("s_wait_asynccnt 0" ::: "memory");
#endif
#endif
}

// ---------------- elementwise f32 -> bf16 convert (4 elems/thread) ----------------
__global__ void __launch_bounds__(256)
cvt_f32_bf16(const float* __restrict__ in, unsigned short* __restrict__ out, long n4) {
  long i = (long)blockIdx.x * blockDim.x + threadIdx.x;
  if (i >= n4) return;
  float4 f = ((const float4*)in)[i];
  uint2 o;
  o.x = (unsigned)f2bf(f.x) | ((unsigned)f2bf(f.y) << 16);
  o.y = (unsigned)f2bf(f.z) | ((unsigned)f2bf(f.w) << 16);
  ((uint2*)out)[i] = o;
}

// ---------------- W[R][C] f32  ->  WT[C][R] bf16 (32x32 LDS tiles) ----------------
__global__ void __launch_bounds__(256)
transpose_cvt(const float* __restrict__ W, unsigned short* __restrict__ WT, int R, int C) {
  __shared__ float tle[32][33];
  int x = blockIdx.x * 32 + threadIdx.x;
  for (int i = 0; i < 4; ++i) {
    int y = blockIdx.y * 32 + threadIdx.y + i * 8;
    tle[threadIdx.y + i * 8][threadIdx.x] = W[(size_t)y * C + x];
  }
  __syncthreads();
  int xo = blockIdx.y * 32 + threadIdx.x;
  for (int i = 0; i < 4; ++i) {
    int yo = blockIdx.x * 32 + threadIdx.y + i * 8;
    WT[(size_t)yo * R + xo] = f2bf(tle[threadIdx.x][threadIdx.y + i * 8]);
  }
}

// ---------------- generic bf16 WMMA GEMM: C[M,N] = A[M,K] * Bt[N,K]^T + bias ------
// Workgroup tile (64*MB) x 64, 4 waves; wave owns MB m-blocks of 16 rows x 64 cols.
// TRANS: store out[n][m] (column-contiguous packing -> 16B stores).
template<bool OUT_BF16, bool TRANS, int MB>
__global__ void __launch_bounds__(128)
gemm_wmma(const unsigned short* __restrict__ A, long Abs,
          const unsigned short* __restrict__ Bt, long Bbs,
          const float* __restrict__ bias,
          void* __restrict__ outv, long Obs,
          int Kd, int ldA, int ldB, int ldOut) {
  __shared__ __align__(16) unsigned short As[64 * MB][40];  // pad 32+8 -> no conflicts
  __shared__ __align__(16) unsigned short Bs[64][40];
  const int tile_n = blockIdx.x * 64;
  const int tile_m = blockIdx.y * 64 * MB;
  A  += (long)blockIdx.z * Abs;
  Bt += (long)blockIdx.z * Bbs;
  const int t = threadIdx.x;
  const int wave = t >> 5, lane = t & 31;
  const int m0 = lane & 15, hl = lane >> 4;
  FragC acc[MB][4];
  for (int mb = 0; mb < MB; ++mb)
    for (int i = 0; i < 4; ++i)
      for (int j = 0; j < 8; ++j) acc[mb][i].f[j] = 0.f;
  for (int kk = 0; kk < Kd; kk += 32) {
    __syncthreads();
    if (MB == 2) {  // A: 128 rows, 1 row/thread; B: 64 rows, half-row/thread
      const unsigned short* ga = &A[(size_t)(tile_m + t) * ldA + kk];
      for (int c = 0; c < 4; ++c) async_ld16(&As[t][c * 8], ga + c * 8);
      const unsigned short* gb = &Bt[(size_t)(tile_n + (t >> 1)) * ldB + kk + (t & 1) * 16];
      for (int c = 0; c < 2; ++c) async_ld16(&Bs[t >> 1][(t & 1) * 16 + c * 8], gb + c * 8);
      __builtin_prefetch(ga + 32, 0, 1);
      __builtin_prefetch(gb + 32, 0, 1);
    } else {        // A,B: 64 rows each, half-row/thread
      const int r = t >> 1, s16 = (t & 1) * 16;
      const unsigned short* ga = &A[(size_t)(tile_m + r) * ldA + kk + s16];
      const unsigned short* gb = &Bt[(size_t)(tile_n + r) * ldB + kk + s16];
      for (int c = 0; c < 2; ++c) async_ld16(&As[r][s16 + c * 8], ga + c * 8);
      for (int c = 0; c < 2; ++c) async_ld16(&Bs[r][s16 + c * 8], gb + c * 8);
      __builtin_prefetch(ga + 32, 0, 1);
      __builtin_prefetch(gb + 32, 0, 1);
    }
    async_wait();
    __syncthreads();
    FragAB a[MB];  // A frag: lane m0 -> row, elements = K {hl*8+0..7, 16+hl*8+0..7}
    for (int mb = 0; mb < MB; ++mb) {
      a[mb].q[0] = *(const uint4*)&As[wave * 16 * MB + mb * 16 + m0][hl * 8];
      a[mb].q[1] = *(const uint4*)&As[wave * 16 * MB + mb * 16 + m0][16 + hl * 8];
    }
    for (int nb = 0; nb < 4; ++nb) {
      FragAB bf;   // B frag: lane m0 -> col n, elements = K {hl*8+0..7, 16+hl*8+0..7}
      bf.q[0] = *(const uint4*)&Bs[nb * 16 + m0][hl * 8];
      bf.q[1] = *(const uint4*)&Bs[nb * 16 + m0][16 + hl * 8];
      for (int mb = 0; mb < MB; ++mb)
        acc[mb][nb].v = __builtin_amdgcn_wmma_f32_16x16x32_bf16(
            false, a[mb].v, false, bf.v, (short)0, acc[mb][nb].v, false, false);
    }
  }
  for (int mb = 0; mb < MB; ++mb) {
    const int mbase = tile_m + wave * 16 * MB + mb * 16 + hl * 8;
    for (int nb = 0; nb < 4; ++nb) {
      int gn = tile_n + nb * 16 + m0;
      float bv = bias ? bias[gn] : 0.f;
      if (TRANS) {
        float vals[8];
        for (int j = 0; j < 8; ++j) vals[j] = acc[mb][nb].f[j] + bv;
        size_t base = (size_t)blockIdx.z * Obs + (size_t)gn * ldOut + mbase;
        if (OUT_BF16) {
          *(uint4*)((unsigned short*)outv + base) = pack8bf(vals);
        } else {
          *(float4*)((float*)outv + base)     = make_float4(vals[0], vals[1], vals[2], vals[3]);
          *(float4*)((float*)outv + base + 4) = make_float4(vals[4], vals[5], vals[6], vals[7]);
        }
      } else {
        for (int j = 0; j < 8; ++j) {
          size_t idx = (size_t)blockIdx.z * Obs + (size_t)(mbase + j) * ldOut + gn;
          float val = acc[mb][nb].f[j] + bv;
          if (OUT_BF16) ((unsigned short*)outv)[idx] = f2bf(val);
          else          ((float*)outv)[idx] = val;
        }
      }
    }
  }
}

// ---------------- flash attention (mask folded out as -1e9 * MVt correction) -----
// grid (S/64, H, B), 4 waves; wave owns 16 q rows. out2[b][h*128+d][q] bf16.
__global__ void __launch_bounds__(128)
mqa_flash(const unsigned short* __restrict__ Q, const unsigned short* __restrict__ K,
          const unsigned short* __restrict__ Vt, const float* __restrict__ MVt,
          unsigned short* __restrict__ out2) {
  constexpr int S = 2048, DH = 128;
  __shared__ __align__(16) unsigned short Ks[64][136];   // [key][d], padded
  __shared__ __align__(16) unsigned short Vts[128][72];  // [d][key], padded
  __shared__ __align__(16) unsigned short Pt[4][64][24]; // per-wave [key][q], padded
  const int qt = blockIdx.x, h = blockIdx.y, b = blockIdx.z;
  const int t = threadIdx.x, wave = t >> 5, lane = t & 31;
  const int m0 = lane & 15, hl = lane >> 4;
  // resident Q A-fragments (16 q rows x 128 d)
  FragAB qf[4];
  {
    const unsigned short* qrow =
        Q + (size_t)(b * S + qt * 64 + wave * 16 + m0) * 2048 + h * DH;
    for (int kc = 0; kc < 4; ++kc) {
      qf[kc].q[0] = *(const uint4*)(qrow + kc * 32 + hl * 8);
      qf[kc].q[1] = *(const uint4*)(qrow + kc * 32 + 16 + hl * 8);
    }
  }
  FragC o[8];
  for (int f = 0; f < 8; ++f)
    for (int j = 0; j < 8; ++j) o[f].f[j] = 0.f;
  float rmax[8], rsum[8];
  for (int j = 0; j < 8; ++j) { rmax[j] = -1e30f; rsum[j] = 0.f; }
  const float scale = 0.088388347648318447f;  // 1/sqrt(128)

  for (int kt = 0; kt < S / 64; ++kt) {
    __syncthreads();
    {  // stage K tile [64][128] and Vt tile [128][64] via async-to-LDS
      int r = t >> 1, seg = (t & 1) * 64;
      const unsigned short* src = K + (size_t)(b * S + kt * 64 + r) * DH + seg;
      for (int c = 0; c < 8; ++c) async_ld16(&Ks[r][seg + c * 8], src + c * 8);
      const unsigned short* vsrc = Vt + ((size_t)b * DH + t) * S + kt * 64;
      for (int c = 0; c < 8; ++c) async_ld16(&Vts[t][c * 8], vsrc + c * 8);
      __builtin_prefetch(src + 64 * DH, 0, 1);   // next K tile
      __builtin_prefetch(vsrc + 64, 0, 1);       // next V tile
    }
    async_wait();
    __syncthreads();
    // scores S = Q.K^T for 16 q x 64 keys
    FragC sf[4];
    for (int nb = 0; nb < 4; ++nb)
      for (int j = 0; j < 8; ++j) sf[nb].f[j] = 0.f;
    for (int kc = 0; kc < 4; ++kc) {
      for (int nb = 0; nb < 4; ++nb) {
        FragAB kb;
        kb.q[0] = *(const uint4*)&Ks[nb * 16 + m0][kc * 32 + hl * 8];
        kb.q[1] = *(const uint4*)&Ks[nb * 16 + m0][kc * 32 + 16 + hl * 8];
        sf[nb].v = __builtin_amdgcn_wmma_f32_16x16x32_bf16(
            false, qf[kc].v, false, kb.v, (short)0, sf[nb].v, false, false);
      }
    }
    // online softmax (rows = hl*8+j; reductions stay within 16-lane halves)
    float corr[8];
    for (int j = 0; j < 8; ++j) {
      float tm = sf[0].f[j] * scale;
      tm = fmaxf(tm, sf[1].f[j] * scale);
      tm = fmaxf(tm, sf[2].f[j] * scale);
      tm = fmaxf(tm, sf[3].f[j] * scale);
      tm = fmaxf(tm, __shfl_xor(tm, 1, 32));
      tm = fmaxf(tm, __shfl_xor(tm, 2, 32));
      tm = fmaxf(tm, __shfl_xor(tm, 4, 32));
      tm = fmaxf(tm, __shfl_xor(tm, 8, 32));
      float nm = fmaxf(rmax[j], tm);
      corr[j] = __expf(rmax[j] - nm);
      rmax[j] = nm;
    }
    float ps[8];
    for (int j = 0; j < 8; ++j) ps[j] = 0.f;
    for (int nb = 0; nb < 4; ++nb) {
      for (int j = 0; j < 8; ++j) {
        float p = __expf(sf[nb].f[j] * scale - rmax[j]);
        ps[j] += p;
        sf[nb].f[j] = p;
      }
      // C-layout column (j) is contiguous -> packed 16B store of P^T[key][q]
      *(uint4*)&Pt[wave][nb * 16 + m0][hl * 8] = pack8bf(sf[nb].f);
    }
    for (int j = 0; j < 8; ++j) {
      ps[j] += __shfl_xor(ps[j], 1, 32);
      ps[j] += __shfl_xor(ps[j], 2, 32);
      ps[j] += __shfl_xor(ps[j], 4, 32);
      ps[j] += __shfl_xor(ps[j], 8, 32);
      rsum[j] = rsum[j] * corr[j] + ps[j];
    }
    for (int f = 0; f < 8; ++f)
      for (int j = 0; j < 8; ++j) o[f].f[j] *= corr[j];
    // O += P @ V  (A = P from wave-private LDS, B = Vt)
    for (int kc = 0; kc < 2; ++kc) {
      FragAB pa;
      for (int e = 0; e < 8; ++e) {
        pa.h[e]     = Pt[wave][kc * 32 + hl * 8 + e][m0];
        pa.h[8 + e] = Pt[wave][kc * 32 + 16 + hl * 8 + e][m0];
      }
      for (int f = 0; f < 8; ++f) {
        FragAB vb;
        vb.q[0] = *(const uint4*)&Vts[f * 16 + m0][kc * 32 + hl * 8];
        vb.q[1] = *(const uint4*)&Vts[f * 16 + m0][kc * 32 + 16 + hl * 8];
        o[f].v = __builtin_amdgcn_wmma_f32_16x16x32_bf16(
            false, pa.v, false, vb.v, (short)0, o[f].v, false, false);
      }
    }
  }
  // epilogue: out2[b][h*128+d][q] = O/l - 1e9*MVt[b][d][q]; packed 16B stores
  const int qcol = qt * 64 + wave * 16 + hl * 8;
  for (int f = 0; f < 8; ++f) {
    int d = f * 16 + m0;
    const float* mv = MVt + ((size_t)b * DH + d) * S + qcol;
    float4 mv0 = *(const float4*)mv;
    float4 mv1 = *(const float4*)(mv + 4);
    float mvv[8] = {mv0.x, mv0.y, mv0.z, mv0.w, mv1.x, mv1.y, mv1.z, mv1.w};
    float vals[8];
    for (int j = 0; j < 8; ++j) vals[j] = o[f].f[j] / rsum[j] - 1e9f * mvv[j];
    *(uint4*)(out2 + ((size_t)b * 2048 + h * DH + d) * S + qcol) = pack8bf(vals);
  }
}

// ---------------------------------- host launch -----------------------------------
extern "C" void kernel_launch(void* const* d_in, const int* in_sizes, int n_in,
                              void* d_out, int out_size, void* d_ws, size_t ws_size,
                              hipStream_t stream) {
  (void)in_sizes; (void)n_in; (void)out_size; (void)ws_size;
  const float* x    = (const float*)d_in[0];
  const float* mask = (const float*)d_in[1];
  const float* Wq = (const float*)d_in[2]; const float* bq = (const float*)d_in[3];
  const float* Wk = (const float*)d_in[4]; const float* bk = (const float*)d_in[5];
  const float* Wv = (const float*)d_in[6]; const float* bv = (const float*)d_in[7];
  const float* Wo = (const float*)d_in[8]; const float* bo = (const float*)d_in[9];

  char* ws = (char*)d_ws;
  unsigned short* xb    = (unsigned short*)(ws + 0);          // [4096][2048] bf16
  unsigned short* maskb = (unsigned short*)(ws + 16777216);   // [2][2048][2048] bf16
  unsigned short* WqT   = (unsigned short*)(ws + 33554432);   // [2048][2048] bf16
  unsigned short* WkT   = (unsigned short*)(ws + 41943040);   // [128][2048] bf16
  unsigned short* WvT   = (unsigned short*)(ws + 42467328);   // [128][2048] bf16
  unsigned short* WoT   = (unsigned short*)(ws + 42991616);   // [2048][2048] bf16
  unsigned short* Qb    = (unsigned short*)(ws + 51380224);   // [4096][2048] bf16
  unsigned short* Kb    = (unsigned short*)(ws + 68157440);   // [4096][128] bf16
  unsigned short* Vtb   = (unsigned short*)(ws + 69206016);   // [2][128][2048] bf16
  float*          MVt   = (float*)         (ws + 70254592);   // [2][128][2048] f32
  unsigned short* out2  = (unsigned short*)(ws + 72351744);   // [2][2048][2048] bf16

  // bf16 conversions of activations + mask
  cvt_f32_bf16<<<8192, 256, 0, stream>>>(x,    xb,    2097152);
  cvt_f32_bf16<<<8192, 256, 0, stream>>>(mask, maskb, 2097152);
  // weight transposes ([in][out] f32 -> [out][in] bf16, for WMMA B-operand)
  dim3 tb(32, 8);
  transpose_cvt<<<dim3(64, 64), tb, 0, stream>>>(Wq, WqT, 2048, 2048);
  transpose_cvt<<<dim3(4, 64),  tb, 0, stream>>>(Wk, WkT, 2048, 128);
  transpose_cvt<<<dim3(4, 64),  tb, 0, stream>>>(Wv, WvT, 2048, 128);
  transpose_cvt<<<dim3(64, 64), tb, 0, stream>>>(Wo, WoT, 2048, 2048);
  // projections (grid.z = batch, M=2048/batch)
  gemm_wmma<true, false, 2><<<dim3(32, 16, 2), 128, 0, stream>>>(
      xb, 2048L * 2048, WqT, 0, bq, Qb, 2048L * 2048, 2048, 2048, 2048, 2048);
  gemm_wmma<true, false, 1><<<dim3(2, 32, 2), 128, 0, stream>>>(
      xb, 2048L * 2048, WkT, 0, bk, Kb, 2048L * 128, 2048, 2048, 2048, 128);
  gemm_wmma<true, true, 1><<<dim3(2, 32, 2), 128, 0, stream>>>(  // V stored transposed
      xb, 2048L * 2048, WvT, 0, bv, Vtb, 128L * 2048, 2048, 2048, 2048, 2048);
  // MVt[b][d][q] = (mask @ V)[q][d]  — head-independent post-softmax mask term
  gemm_wmma<false, true, 1><<<dim3(2, 32, 2), 128, 0, stream>>>(
      maskb, 2048L * 2048, Vtb, 128L * 2048, nullptr, MVt, 128L * 2048,
      2048, 2048, 2048, 2048);
  // flash attention -> out2 (quirky [h*128+d][q] layout for the final GEMM)
  mqa_flash<<<dim3(32, 16, 2), 128, 0, stream>>>(Qb, Kb, Vtb, MVt, out2);
  // final = out2 @ Wo + bo  (contracts over sequence axis), fp32 out
  gemm_wmma<false, false, 2><<<dim3(32, 16, 2), 128, 0, stream>>>(
      out2, 2048L * 2048, WoT, 0, bo, (float*)d_out, 2048L * 2048,
      2048, 2048, 2048, 2048);
}
